// YOLOV3_25151328485598
// MI455X (gfx1250) — 8-bit matrix operand order repaired and checked by reading, compiled, not run on hardware
//
#include <hip/hip_runtime.h>
#include <hip/hip_bf16.h>

#define AA 3
#define CC 80
#define BB 8
static constexpr int H0_ = 19, H1_ = 38, H2_ = 76;
static constexpr int N0_ = H0_ * H0_ * AA;
static constexpr int N1_ = H1_ * H1_ * AA;
static constexpr int N2_ = H2_ * H2_ * AA;
static constexpr int NTOT_ = N0_ + N1_ + N2_;
static constexpr int NC_ = NTOT_ * CC;
static constexpr int NBINS_ = 4096;
static constexpr int CAP_ = 8192;
static constexpr int KTOP_ = 400;
static constexpr int POST_ = 100;
static constexpr float NMSTHR_ = 0.45f;
static constexpr int NW_ = 13;

typedef __attribute__((ext_vector_type(16))) int   v16i;
typedef __attribute__((ext_vector_type(8)))  float v8f;

__device__ __forceinline__ float sigm(float x) { return 1.0f / (1.0f + expf(-x)); }

__device__ __forceinline__ unsigned keyOf(float f) {
    unsigned u = __float_as_uint(f);
    return (u & 0x80000000u) ? ~u : (u | 0x80000000u);
}
__device__ __forceinline__ float keyInv(unsigned k) {
    unsigned u = (k & 0x80000000u) ? (k & 0x7FFFFFFFu) : ~k;
    return __uint_as_float(u);
}

__device__ __forceinline__ void locate(int n, int& s, int& nl, int& W, float& stride) {
    if (n < N0_)              { s = 0; nl = n;               W = H0_; stride = 32.f; }
    else if (n < N0_ + N1_)   { s = 1; nl = n - N0_;         W = H1_; stride = 16.f; }
    else                      { s = 2; nl = n - (N0_ + N1_); W = H2_; stride = 8.f;  }
}

__device__ __forceinline__ float candScore(int b, int r,
        const float* __restrict__ t0, const float* __restrict__ t1,
        const float* __restrict__ t2, const float* __restrict__ conf) {
    int n = r / CC;
    int c = r - n * CC;
    int s, nl, W; float stride;
    locate(n, s, nl, W, stride);
    const float* tip = (s == 0) ? t0 : ((s == 1) ? t1 : t2);
    int a  = nl % AA;
    int hw = nl / AA;
    int ch = a * (CC + 5) + 5 + c;
    size_t plane = (size_t)W * W;
    float logit = tip[((size_t)b * (AA * (CC + 5)) + ch) * plane + hw];
    return sigm(logit) * conf[b * NTOT_ + n];
}

__global__ void yolo_zero_kernel(unsigned* hist, unsigned* candcnt) {
    int i = blockIdx.x * blockDim.x + threadIdx.x;
    if (i < BB * NBINS_) hist[i] = 0u;
    if (i < BB) candcnt[i] = 0u;
}

__global__ void yolo_decode_kernel(const float* __restrict__ t0, const float* __restrict__ t1,
                                   const float* __restrict__ t2, const float* __restrict__ a0,
                                   const float* __restrict__ a1, const float* __restrict__ a2,
                                   float* __restrict__ bbox, float* __restrict__ conf) {
    int idx = blockIdx.x * blockDim.x + threadIdx.x;
    if (idx >= BB * NTOT_) return;
    int b = idx / NTOT_, n = idx - b * NTOT_;
    int s, nl, W; float stride;
    locate(n, s, nl, W, stride);
    const float* tip = (s == 0) ? t0 : ((s == 1) ? t1 : t2);
    const float* anc = (s == 0) ? a0 : ((s == 1) ? a1 : a2);
    int a  = nl % AA;
    int hw = nl / AA;
    int gy = hw / W, gx = hw - gy * W;
    size_t plane = (size_t)W * W;
    size_t base = ((size_t)b * (AA * (CC + 5)) + a * (CC + 5)) * plane + hw;
    float p0 = tip[base];
    float p1 = tip[base + plane];
    float p2 = tip[base + 2 * plane];
    float p3 = tip[base + 3 * plane];
    float p4 = tip[base + 4 * plane];
    float cx = (sigm(p0) + (float)gx) * stride;
    float cy = (sigm(p1) + (float)gy) * stride;
    float hwd = expf(p2) * anc[a * 2 + 0] * 0.5f;
    float hhd = expf(p3) * anc[a * 2 + 1] * 0.5f;
    float* o = bbox + (size_t)idx * 4;
    o[0] = cx - hwd; o[1] = cy - hhd; o[2] = cx + hwd; o[3] = cy + hhd;
    conf[idx] = sigm(p4);
}

__global__ void yolo_hist_kernel(const float* __restrict__ t0, const float* __restrict__ t1,
                                 const float* __restrict__ t2, const float* __restrict__ conf,
                                 unsigned* __restrict__ hist) {
    int idx = blockIdx.x * blockDim.x + threadIdx.x;
    if (idx >= BB * NC_) return;
    int b = idx / NC_, r = idx - b * NC_;
    float sc = candScore(b, r, t0, t1, t2, conf);
    unsigned k = keyOf(sc);
    atomicAdd(&hist[b * NBINS_ + (k >> 20)], 1u);
}

__global__ void yolo_thresh_kernel(const unsigned* __restrict__ hist, unsigned* __restrict__ binthr) {
    int b = blockIdx.x * blockDim.x + threadIdx.x;
    if (b >= BB) return;
    unsigned acc = 0;
    int bin = NBINS_ - 1;
    for (; bin > 0; --bin) {
        acc += hist[b * NBINS_ + bin];
        if (acc >= (unsigned)KTOP_) break;
    }
    binthr[b] = (unsigned)bin;
}

__global__ void yolo_collect_kernel(const float* __restrict__ t0, const float* __restrict__ t1,
                                    const float* __restrict__ t2, const float* __restrict__ conf,
                                    const unsigned* __restrict__ binthr,
                                    unsigned long long* __restrict__ cand,
                                    unsigned* __restrict__ candcnt) {
    int idx = blockIdx.x * blockDim.x + threadIdx.x;
    if (idx >= BB * NC_) return;
    int b = idx / NC_, r = idx - b * NC_;
    float sc = candScore(b, r, t0, t1, t2, conf);
    unsigned k = keyOf(sc);
    if ((k >> 20) >= binthr[b]) {
        unsigned pos = atomicAdd(&candcnt[b], 1u);
        if (pos < (unsigned)CAP_) {
            cand[(size_t)b * CAP_ + pos] =
                ((unsigned long long)k << 32) | (unsigned long long)(0xFFFFFFFFu - (unsigned)r);
        }
    }
}

__global__ __launch_bounds__(1024) void yolo_topk_kernel(const unsigned long long* __restrict__ cand,
                                                         const unsigned* __restrict__ candcnt,
                                                         const float* __restrict__ bbox,
                                                         float* __restrict__ det) {
    __shared__ unsigned long long arr[CAP_];
    int b = blockIdx.x, tid = threadIdx.x;
    unsigned cnt = candcnt[b];
    if (cnt > (unsigned)CAP_) cnt = CAP_;
    for (int i = tid; i < CAP_; i += 1024)
        arr[i] = (i < (int)cnt) ? cand[(size_t)b * CAP_ + i] : 0ull;
    __syncthreads();
    for (int size = 2; size <= CAP_; size <<= 1) {
        for (int str = size >> 1; str > 0; str >>= 1) {
            for (int i = tid; i < CAP_; i += 1024) {
                int j = i ^ str;
                if (j > i) {
                    unsigned long long x = arr[i], y = arr[j];
                    bool descSeg = ((i & size) == 0);
                    bool doSwap = descSeg ? (x < y) : (x > y);
                    if (doSwap) { arr[i] = y; arr[j] = x; }
                }
            }
            __syncthreads();
        }
    }
    if (tid < KTOP_) {
        unsigned long long v = arr[tid];
        unsigned k = (unsigned)(v >> 32);
        unsigned r = 0xFFFFFFFFu - (unsigned)(v & 0xFFFFFFFFull);
        float sc = keyInv(k);
        int n = (int)(r / CC);
        int c = (int)(r - (unsigned)n * CC);
        float* row = det + ((size_t)b * KTOP_ + tid) * 6;
        const float* bx = bbox + ((size_t)b * NTOT_ + n) * 4;
        bool real = (v != 0ull);
        row[0] = real ? (float)c : -1.f;
        row[1] = real ? sc : -1.f;
        row[2] = real ? bx[0] : -1.f;
        row[3] = real ? bx[1] : -1.f;
        row[4] = real ? bx[2] : -1.f;
        row[5] = real ? bx[3] : -1.f;
    }
}

__global__ __launch_bounds__(512) void yolo_nms_kernel(const float* __restrict__ det,
                                                       float* __restrict__ out) {
    __shared__ float sx1[KTOP_], sy1[KTOP_], sx2[KTOP_], sy2[KTOP_];
    __shared__ float sScore[KTOP_], sArea[KTOP_];
    __shared__ int   sCid[KTOP_];
    __shared__ unsigned supMat[KTOP_ * NW_];
    __shared__ unsigned keepW[NW_];
    __shared__ int outIdx[POST_];

    int b = blockIdx.x, tid = threadIdx.x;
    for (int i = tid; i < KTOP_; i += 512) {
        const float* row = det + ((size_t)b * KTOP_ + i) * 6;
        sCid[i] = (int)row[0];
        sScore[i] = row[1];
        float x1 = row[2], y1 = row[3], x2 = row[4], y2 = row[5];
        sx1[i] = x1; sy1[i] = y1; sx2[i] = x2; sy2[i] = y2;
        sArea[i] = fmaxf(x2 - x1, 0.f) * fmaxf(y2 - y1, 0.f);
    }
    for (int i = tid; i < KTOP_ * NW_; i += 512) supMat[i] = 0u;
    __syncthreads();

    int wave = tid >> 5;
    int lane = (int)(tid & 31u);
    int laneM = lane & 15;
    int half  = lane >> 4;
    v8f cz = {0.f, 0.f, 0.f, 0.f, 0.f, 0.f, 0.f, 0.f};
    for (int t = wave; t < 25 * 25; t += 16) {
        int ti = t / 25, tj = t - ti * 25;
        int cidA = sCid[ti * 16 + laneM];
        int cidB = sCid[tj * 16 + laneM];
        int va = ((cidA >> 5) << 2) | ((cidA >> 2) & 3);
        int wa = (((cidA >> 4) & 1) == half) ? (0x38 << ((cidA & 3) * 8)) : 0;
        int vb = ((cidB >> 5) << 2) | ((cidB >> 2) & 3);
        int wb = (((cidB >> 4) & 1) == half) ? (0x38 << ((cidB & 3) * 8)) : 0;
        v16i av, bv;
#pragma unroll
        for (int v = 0; v < 16; ++v) {
            av[v] = (v == va) ? wa : 0;
            bv[v] = (v == vb) ? wb : 0;
        }
        v8f d = __builtin_amdgcn_wmma_f32_16x16x128_fp8_fp8(av, bv, (short)0, cz, false, false);
#pragma unroll
        for (int r = 0; r < 8; ++r) {
            int i  = ti * 16 + r + (half << 3);
            int jg = tj * 16 + laneM;
            if (d[r] > 0.5f && jg > i) {
                float iw = fminf(sx2[i], sx2[jg]) - fmaxf(sx1[i], sx1[jg]);
                float ih = fminf(sy2[i], sy2[jg]) - fmaxf(sy1[i], sy1[jg]);
                iw = fmaxf(iw, 0.f); ih = fmaxf(ih, 0.f);
                float inter = iw * ih;
                float iou = inter / (sArea[i] + sArea[jg] - inter + 1e-12f);
                if (iou > NMSTHR_)
                    atomicOr(&supMat[i * NW_ + (jg >> 5)], 1u << (jg & 31));
            }
        }
    }
    __syncthreads();

    if (tid == 0) {
        for (int w = 0; w < NW_; ++w) keepW[w] = (w == NW_ - 1) ? 0x0000FFFFu : 0xFFFFFFFFu;
        for (int i = 0; i < KTOP_; ++i) {
            if (((keepW[i >> 5] >> (i & 31)) & 1u) && sScore[i] > 0.f) {
                for (int w = 0; w < NW_; ++w) keepW[w] &= ~supMat[i * NW_ + w];
            }
        }
        int cnt = 0;
        for (int i = 0; i < KTOP_ && cnt < POST_; ++i)
            if (((keepW[i >> 5] >> (i & 31)) & 1u) && sScore[i] > 0.f) outIdx[cnt++] = i;
        for (; cnt < POST_; ++cnt) outIdx[cnt] = -1;
    }
    __syncthreads();

    float* ids = out;
    float* scs = out + BB * POST_;
    float* bbs = out + 2 * BB * POST_;
    for (int k = tid; k < POST_; k += 512) {
        int i = outIdx[k];
        int o = b * POST_ + k;
        if (i >= 0) {
            ids[o] = (float)sCid[i];
            scs[o] = sScore[i];
            bbs[(size_t)o * 4 + 0] = sx1[i];
            bbs[(size_t)o * 4 + 1] = sy1[i];
            bbs[(size_t)o * 4 + 2] = sx2[i];
            bbs[(size_t)o * 4 + 3] = sy2[i];
        } else {
            ids[o] = -1.f; scs[o] = -1.f;
            bbs[(size_t)o * 4 + 0] = -1.f;
            bbs[(size_t)o * 4 + 1] = -1.f;
            bbs[(size_t)o * 4 + 2] = -1.f;
            bbs[(size_t)o * 4 + 3] = -1.f;
        }
    }
}

extern "C" void kernel_launch(void* const* d_in, const int* in_sizes, int n_in,
                              void* d_out, int out_size, void* d_ws, size_t ws_size,
                              hipStream_t stream) {
    const float* t0 = (const float*)d_in[0];
    const float* t1 = (const float*)d_in[1];
    const float* t2 = (const float*)d_in[2];
    const float* a0 = (const float*)d_in[3];
    const float* a1 = (const float*)d_in[4];
    const float* a2 = (const float*)d_in[5];

    float* bbox = (float*)d_ws;
    float* conf = bbox + (size_t)BB * NTOT_ * 4;
    unsigned* hist = (unsigned*)(conf + (size_t)BB * NTOT_);
    unsigned* binthr = hist + (size_t)BB * NBINS_;
    unsigned* candcnt = binthr + BB;
    unsigned long long* cand = (unsigned long long*)(candcnt + BB);
    float* det = (float*)(cand + (size_t)BB * CAP_);

    float* out = (float*)d_out;

    yolo_zero_kernel<<<(BB * NBINS_ + 255) / 256, 256, 0, stream>>>(hist, candcnt);
    yolo_decode_kernel<<<(BB * NTOT_ + 255) / 256, 256, 0, stream>>>(t0, t1, t2, a0, a1, a2,
                                                                     bbox, conf);
    yolo_hist_kernel<<<(BB * NC_ + 255) / 256, 256, 0, stream>>>(t0, t1, t2, conf, hist);
    yolo_thresh_kernel<<<1, BB, 0, stream>>>(hist, binthr);
    yolo_collect_kernel<<<(BB * NC_ + 255) / 256, 256, 0, stream>>>(t0, t1, t2, conf, binthr,
                                                                    cand, candcnt);
    yolo_topk_kernel<<<BB, 1024, 0, stream>>>(cand, candcnt, bbox, det);
    yolo_nms_kernel<<<BB, 512, 0, stream>>>(det, out);
}
